// SequenceTower_84782654423225
// MI455X (gfx1250) — compile-verified, matrix operand order checked by reading, not run on hardware
//
#include <hip/hip_runtime.h>
#include <hip/hip_bf16.h>
#include <stdint.h>

// ---------------- problem constants ----------------
#define L_SEQ   16384
#define DIN     256
#define DMODEL  768
#define NLAYER  8
#define EPSF    1e-5f
#define CHUNK   128
#define NCHUNK  (L_SEQ / CHUNK)   // 128

typedef unsigned short bf_t;   // bf16 storage bits
typedef __attribute__((ext_vector_type(16))) __bf16 v16bf;
typedef __attribute__((ext_vector_type(8)))  float  v8f;
typedef __attribute__((ext_vector_type(4)))  int    v4i_;

union FragBF { v16bf v; uint4 q[2]; };

__device__ __forceinline__ bf_t f2bf(float f) {
  unsigned u = __float_as_uint(f);
  unsigned r = u + 0x7FFFu + ((u >> 16) & 1u);   // round-to-nearest-even
  return (bf_t)(r >> 16);
}
__device__ __forceinline__ float sigmoidf_(float x) { return 1.0f / (1.0f + __expf(-x)); }

// ---- CDNA5 async global->LDS path (guarded; falls back to manual copies) ----
#if defined(__has_builtin)
#if __has_builtin(__builtin_amdgcn_global_load_async_to_lds_b128) && \
    __has_builtin(__builtin_amdgcn_s_wait_asynccnt)
#define ST_ASYNC 1
#endif
#endif

typedef __attribute__((address_space(1))) v4i_ g_v4i;   // global int4
typedef __attribute__((address_space(3))) v4i_ l_v4i;   // LDS int4

// ---------------- f32 -> bf16 convert (row-major, for activations) ----------------
__global__ void st_cvt_bf16(const float* __restrict__ in, bf_t* __restrict__ out, int n) {
  int i = blockIdx.x * blockDim.x + threadIdx.x;
  if (i < n) out[i] = f2bf(in[i]);
}

// -------- f32 [K,N] -> bf16 transposed [N,K] (weights; batched over layers) -------
__global__ __launch_bounds__(256)
void st_cvt_bf16_t(const float* __restrict__ in, bf_t* __restrict__ out, int K, int N) {
  __shared__ bf_t tile[32][33];
  const size_t zin  = (size_t)blockIdx.z * K * N;
  const int n0 = blockIdx.x * 32, k0 = blockIdx.y * 32;
  const int tx = threadIdx.x & 31, ty = threadIdx.x >> 5;   // 32 x 8
  #pragma unroll
  for (int i = ty; i < 32; i += 8)
    tile[i][tx] = f2bf(in[zin + (size_t)(k0 + i) * N + n0 + tx]);
  __syncthreads();
  #pragma unroll
  for (int i = ty; i < 32; i += 8)
    out[zin + (size_t)(n0 + i) * K + k0 + tx] = tile[tx][i];
}

// ---------------- WMMA GEMM ----------------------------------------------------
// C[M,N] = A[M,K](bf16,row-major) * Bt[N,K](bf16, pre-transposed) + bias, epilogue.
// 256 threads = 8 waves in a 4(row)x2(col) grid; BM=128, BN=128, BK=32.
// Each wave: 2 A frags x 4 B frags -> 8 v_wmma_f32_16x16x32_bf16 per K-step.
// Double-buffered LDS, async global->LDS DMA overlapped with WMMA.
#define EPI_BIAS 0
#define EPI_RES  1
#define EPI_SIG  2
#define EPI_SILU 3

template<int EPI, typename OutT>
__global__ __launch_bounds__(256)
void st_gemm_wmma(const bf_t* __restrict__ A, const bf_t* __restrict__ Bt,
                  const float* __restrict__ bias, const float* __restrict__ res,
                  OutT* __restrict__ C, int M, int N, int K) {
  __shared__ __align__(16) bf_t sA[2][128 * 40];   // 128 rows x 32 K (+8 pad)
  __shared__ __align__(16) bf_t sB[2][128 * 40];   // 128 cols x 32 K (+8 pad)

  const int tid  = threadIdx.x;
  const int lane = tid & 31;
  const int wid  = tid >> 5;
  const int wrow = wid & 3;          // 4 row-waves  : rows  wrow*32 .. +32
  const int wcol = wid >> 2;         // 2 col-waves  : cols  wcol*64 .. +64
  const int bm = blockIdx.y * 128;
  const int bn = blockIdx.x * 128;

  v8f acc[2][4];
  #pragma unroll
  for (int i = 0; i < 2; ++i)
    #pragma unroll
    for (int j = 0; j < 4; ++j)
      #pragma unroll
      for (int r = 0; r < 8; ++r) acc[i][j][r] = 0.0f;

  // staging map: 256 threads; thread copies 32B of A and 32B of Bt per K-step
  const int crow = tid >> 1, chalf = tid & 1;
  const bf_t* gA = A  + (size_t)(bm + crow) * K + chalf * 16;
  const bf_t* gB = Bt + (size_t)(bn + crow) * K + chalf * 16;
  const int sOff = crow * 40 + chalf * 16;

  auto stage = [&](int b, int k0) {
#if ST_ASYNC
    __builtin_amdgcn_global_load_async_to_lds_b128(
        (g_v4i*)(gA + k0),     (l_v4i*)&sA[b][sOff],     0, 0);
    __builtin_amdgcn_global_load_async_to_lds_b128(
        (g_v4i*)(gA + k0 + 8), (l_v4i*)&sA[b][sOff + 8], 0, 0);
    __builtin_amdgcn_global_load_async_to_lds_b128(
        (g_v4i*)(gB + k0),     (l_v4i*)&sB[b][sOff],     0, 0);
    __builtin_amdgcn_global_load_async_to_lds_b128(
        (g_v4i*)(gB + k0 + 8), (l_v4i*)&sB[b][sOff + 8], 0, 0);
#else
    const uint4* ga = (const uint4*)(gA + k0);
    const uint4* gb = (const uint4*)(gB + k0);
    uint4* da = (uint4*)&sA[b][sOff];
    uint4* db = (uint4*)&sB[b][sOff];
    da[0] = ga[0]; da[1] = ga[1];
    db[0] = gb[0]; db[1] = gb[1];
    if (k0 + 32 < K) __builtin_prefetch(gA + k0 + 32, 0, 0);
#endif
  };

  const int m16 = lane & 15, kh = lane >> 4;
  const int aoBase = (wrow * 32 + m16) * 40 + kh * 16;
  const int boBase = (wcol * 64 + m16) * 40 + kh * 16;

  stage(0, 0);
  int buf = 0;
  for (int k0 = 0; k0 < K; k0 += 32) {
#if ST_ASYNC
    __builtin_amdgcn_s_wait_asynccnt(0);
#endif
    __syncthreads();                       // buf ready; prev reads of buf^1 retired
    if (k0 + 32 < K) stage(buf ^ 1, k0 + 32);

    FragBF fa0, fa1;
    fa0.q[0] = *(const uint4*)&sA[buf][aoBase];
    fa0.q[1] = *(const uint4*)&sA[buf][aoBase + 8];
    fa1.q[0] = *(const uint4*)&sA[buf][aoBase + 16 * 40];
    fa1.q[1] = *(const uint4*)&sA[buf][aoBase + 16 * 40 + 8];
    #pragma unroll
    for (int j = 0; j < 4; ++j) {
      FragBF fb;
      const int bo = boBase + j * 16 * 40;
      fb.q[0] = *(const uint4*)&sB[buf][bo];
      fb.q[1] = *(const uint4*)&sB[buf][bo + 8];
      acc[0][j] = __builtin_amdgcn_wmma_f32_16x16x32_bf16(false, fa0.v, false, fb.v,
                                                          (short)0, acc[0][j], false, false);
      acc[1][j] = __builtin_amdgcn_wmma_f32_16x16x32_bf16(false, fa1.v, false, fb.v,
                                                          (short)0, acc[1][j], false, false);
    }
    buf ^= 1;
  }

  // epilogue: lanes 0-15 -> M=r, lanes 16-31 -> M=r+8; N = lane&15
  const int rb = bm + wrow * 32 + (lane >> 4) * 8;
  const int cb = bn + wcol * 64 + (lane & 15);
  #pragma unroll
  for (int i = 0; i < 2; ++i) {
    #pragma unroll
    for (int j = 0; j < 4; ++j) {
      const int col = cb + j * 16;
      const float bv = bias[col];
      #pragma unroll
      for (int r = 0; r < 8; ++r) {
        const int row = rb + i * 16 + r;
        const size_t off = (size_t)row * N + col;
        float v = acc[i][j][r] + bv;
        if (EPI == EPI_SIG)  v = sigmoidf_(v);
        if (EPI == EPI_SILU) v = v * sigmoidf_(v);
        if (EPI == EPI_RES)  v += res[off];
        if constexpr (sizeof(OutT) == 2) ((bf_t*)C)[off] = f2bf(v);
        else                              ((float*)C)[off] = v;
      }
    }
  }
}

// ---------------- LayerNorm over D=768, one row per block ----------------
template<typename OutT>
__global__ __launch_bounds__(256)
void st_layernorm(const float* __restrict__ x, const float* __restrict__ g,
                  const float* __restrict__ b, OutT* __restrict__ out) {
  const int row = blockIdx.x;
  const int tid = threadIdx.x;
  const float* xr = x + (size_t)row * DMODEL;
  const float v0 = xr[tid], v1 = xr[tid + 256], v2 = xr[tid + 512];
  float s = v0 + v1 + v2;
  float q = v0 * v0 + v1 * v1 + v2 * v2;
  #pragma unroll
  for (int o = 16; o > 0; o >>= 1) {
    s += __shfl_down(s, o, 32);
    q += __shfl_down(q, o, 32);
  }
  __shared__ float rs[8], rq[8];
  if ((tid & 31) == 0) { rs[tid >> 5] = s; rq[tid >> 5] = q; }
  __syncthreads();
  float ts = 0.f, tq = 0.f;
  #pragma unroll
  for (int i = 0; i < 8; ++i) { ts += rs[i]; tq += rq[i]; }
  const float mu   = ts * (1.0f / DMODEL);
  const float var  = tq * (1.0f / DMODEL) - mu * mu;
  const float rstd = rsqrtf(var + EPSF);
  #pragma unroll
  for (int i = 0; i < 3; ++i) {
    const int d = tid + i * 256;
    const float y = (xr[d] - mu) * rstd * g[d] + b[d];
    if constexpr (sizeof(OutT) == 2) ((bf_t*)out)[(size_t)row * DMODEL + d] = f2bf(y);
    else                              ((float*)out)[(size_t)row * DMODEL + d] = y;
  }
}

// ---------------- 3-phase blocked scan: state = decay*state + u_t ----------------
__global__ __launch_bounds__(256)
void st_scan_chunk(float* __restrict__ u, const float* __restrict__ dlogit,
                   float* __restrict__ carry) {
  const int chunk = blockIdx.x;
  const int tid = threadIdx.x;
  float d[3], st[3];
  #pragma unroll
  for (int i = 0; i < 3; ++i) { d[i] = sigmoidf_(dlogit[tid + i * 256]); st[i] = 0.f; }
  const size_t base = (size_t)chunk * CHUNK * DMODEL + tid;
  for (int t = 0; t < CHUNK; ++t) {
    #pragma unroll
    for (int i = 0; i < 3; ++i) {
      const size_t idx = base + (size_t)t * DMODEL + i * 256;
      st[i] = d[i] * st[i] + u[idx];
      u[idx] = st[i];
    }
  }
  #pragma unroll
  for (int i = 0; i < 3; ++i) carry[(size_t)chunk * DMODEL + tid + i * 256] = st[i];
}

__global__ __launch_bounds__(768)
void st_scan_carry(const float* __restrict__ carry, const float* __restrict__ dlogit,
                   float* __restrict__ prefix) {
  const int ch = threadIdx.x;
  const float d  = sigmoidf_(dlogit[ch]);
  const float dP = __powf(d, (float)CHUNK);
  float run = 0.f;
  for (int c = 0; c < NCHUNK; ++c) {
    run = dP * run + carry[(size_t)c * DMODEL + ch];
    prefix[(size_t)c * DMODEL + ch] = run;
  }
}

__global__ __launch_bounds__(256)
void st_scan_fix(const float* __restrict__ states, const float* __restrict__ prefix,
                 const float* __restrict__ gate, const float* __restrict__ dlogit,
                 bf_t* __restrict__ y) {
  const size_t i = (size_t)blockIdx.x * blockDim.x + threadIdx.x;
  const int ch = (int)(i % DMODEL);
  const size_t t = i / DMODEL;
  const int chunk = (int)(t >> 7);
  const int tin   = (int)(t & (CHUNK - 1));
  float s = states[i];
  if (chunk > 0) {
    const float d = sigmoidf_(dlogit[ch]);
    s += prefix[(size_t)(chunk - 1) * DMODEL + ch] * __powf(d, (float)(tin + 1));
  }
  y[i] = f2bf(s * gate[i]);
}

// ---------------- host orchestration ----------------
extern "C" void kernel_launch(void* const* d_in, const int* in_sizes, int n_in,
                              void* d_out, int out_size, void* d_ws, size_t ws_size,
                              hipStream_t stream) {
  (void)in_sizes; (void)n_in; (void)out_size; (void)ws_size;
  const float* node     = (const float*)d_in[0];
  const float* Wproj    = (const float*)d_in[1];
  const float* bproj    = (const float*)d_in[2];
  const float* ln_s_g   = (const float*)d_in[3];
  const float* ln_s_b   = (const float*)d_in[4];
  const float* W_in     = (const float*)d_in[5];
  const float* b_in     = (const float*)d_in[6];
  const float* W_gate   = (const float*)d_in[7];
  const float* b_gate   = (const float*)d_in[8];
  const float* W_out    = (const float*)d_in[9];
  const float* b_out    = (const float*)d_in[10];
  const float* dlogit   = (const float*)d_in[11];
  const float* ln_f_g   = (const float*)d_in[12];
  const float* ln_f_b   = (const float*)d_in[13];
  const float* W_ff1    = (const float*)d_in[14];
  const float* b_ff1    = (const float*)d_in[15];
  const float* W_ff2    = (const float*)d_in[16];
  const float* b_ff2    = (const float*)d_in[17];
  const float* ln_o_g   = (const float*)d_in[18];
  const float* ln_o_b   = (const float*)d_in[19];
  float* out = (float*)d_out;

  const size_t LD = (size_t)L_SEQ * DMODEL;
  const size_t DD = (size_t)DMODEL * DMODEL;
  const size_t D2 = (size_t)DMODEL * 2 * DMODEL;

  size_t off = 0;
  auto carve = [&](size_t bytes) -> char* {
    char* p = (char*)d_ws + off;
    off += (bytes + 255) & ~(size_t)255;
    return p;
  };
  float* x       = (float*)carve(LD * 4);
  float* u       = (float*)carve(LD * 4);            // reused in-place as scan states
  float* gate    = (float*)carve(LD * 4);
  bf_t*  h_bf    = (bf_t*) carve(LD * 2);
  bf_t*  y_bf    = (bf_t*) carve(LD * 2);
  bf_t*  f1_bf   = (bf_t*) carve((size_t)L_SEQ * 2 * DMODEL * 2);
  bf_t*  node_bf = (bf_t*) carve((size_t)L_SEQ * DIN * 2);
  float* carry   = (float*)carve((size_t)NCHUNK * DMODEL * 4);
  float* prefix  = (float*)carve((size_t)NCHUNK * DMODEL * 4);
  bf_t*  wproj_t = (bf_t*)carve((size_t)DIN * DMODEL * 2);   // [D, DIN]
  bf_t*  win_t   = (bf_t*)carve(NLAYER * DD * 2);            // [D, D] per layer
  bf_t*  wgate_t = (bf_t*)carve(NLAYER * DD * 2);
  bf_t*  wout_t  = (bf_t*)carve(NLAYER * DD * 2);
  bf_t*  wff1_t  = (bf_t*)carve(NLAYER * D2 * 2);            // [2D, D] per layer
  bf_t*  wff2_t  = (bf_t*)carve(NLAYER * D2 * 2);            // [D, 2D] per layer

  // activations: plain convert; weights: convert + transpose to [N,K]
  st_cvt_bf16<<<(unsigned)((L_SEQ * DIN + 255) / 256), 256, 0, stream>>>(
      node, node_bf, L_SEQ * DIN);
  auto cvtT = [&](const float* src, bf_t* dst, int K, int N, int batch) {
    st_cvt_bf16_t<<<dim3(N / 32, K / 32, batch), 256, 0, stream>>>(src, dst, K, N);
  };
  cvtT(Wproj,  wproj_t, DIN,        DMODEL,     1);
  cvtT(W_in,   win_t,   DMODEL,     DMODEL,     NLAYER);
  cvtT(W_gate, wgate_t, DMODEL,     DMODEL,     NLAYER);
  cvtT(W_out,  wout_t,  DMODEL,     DMODEL,     NLAYER);
  cvtT(W_ff1,  wff1_t,  DMODEL,     2 * DMODEL, NLAYER);
  cvtT(W_ff2,  wff2_t,  2 * DMODEL, DMODEL,     NLAYER);

  const dim3 gD (DMODEL / 128,     L_SEQ / 128);   // N=768  -> (6,128)
  const dim3 g2D(2 * DMODEL / 128, L_SEQ / 128);   // N=1536 -> (12,128)

  // x = node @ Wproj + bproj
  st_gemm_wmma<EPI_BIAS, float><<<gD, 256, 0, stream>>>(
      node_bf, wproj_t, bproj, nullptr, x, L_SEQ, DMODEL, DIN);

  const unsigned fixBlocks = (unsigned)(LD / 256);
  for (int l = 0; l < NLAYER; ++l) {
    const size_t lD = (size_t)l * DMODEL;
    st_layernorm<bf_t><<<L_SEQ, 256, 0, stream>>>(x, ln_s_g + lD, ln_s_b + lD, h_bf);
    st_gemm_wmma<EPI_BIAS, float><<<gD, 256, 0, stream>>>(
        h_bf, win_t + l * DD, b_in + lD, nullptr, u, L_SEQ, DMODEL, DMODEL);
    st_gemm_wmma<EPI_SIG, float><<<gD, 256, 0, stream>>>(
        h_bf, wgate_t + l * DD, b_gate + lD, nullptr, gate, L_SEQ, DMODEL, DMODEL);
    st_scan_chunk<<<NCHUNK, 256, 0, stream>>>(u, dlogit + lD, carry);
    st_scan_carry<<<1, DMODEL, 0, stream>>>(carry, dlogit + lD, prefix);
    st_scan_fix<<<fixBlocks, 256, 0, stream>>>(u, prefix, gate, dlogit + lD, y_bf);
    st_gemm_wmma<EPI_RES, float><<<gD, 256, 0, stream>>>(
        y_bf, wout_t + l * DD, b_out + lD, x, x, L_SEQ, DMODEL, DMODEL);
    st_layernorm<bf_t><<<L_SEQ, 256, 0, stream>>>(x, ln_f_g + lD, ln_f_b + lD, h_bf);
    st_gemm_wmma<EPI_SILU, bf_t><<<g2D, 256, 0, stream>>>(
        h_bf, wff1_t + l * D2, b_ff1 + (size_t)l * 2 * DMODEL, nullptr, f1_bf,
        L_SEQ, 2 * DMODEL, DMODEL);
    st_gemm_wmma<EPI_RES, float><<<gD, 256, 0, stream>>>(
        f1_bf, wff2_t + l * D2, b_ff2 + lD, x, x, L_SEQ, DMODEL, 2 * DMODEL);
  }
  st_layernorm<float><<<L_SEQ, 256, 0, stream>>>(x, ln_o_g, ln_o_b, out);
}
